// Module_RayCasting_85272280694872
// MI455X (gfx1250) — matrix operand order repaired and checked by reading, not compiled, not run on hardware
//
#include <hip/hip_runtime.h>
#include <math.h>

// ---------------------------------------------------------------------------
// Neural implicit ray casting for MI455X (gfx1250, wave32, WMMA).
//
// Decoder: h1 = tanh(p @ W1 + b1); h2 = tanh(h1 @ W2 + b2); f = h2 @ W3 + b3
// Dominant cost: 256x256 layer-2 GEMM -> V_WMMA_F32_16X16X32_F16 (f32 accum).
// W2 kept f16 column-major, resident in LDS (320KB/WGP on CDNA5).
// 512-thread blocks (16 waves) amortize the W2T LDS footprint -> 4 waves/SIMD.
// B tiles batch-loaded per N-tile (16x ds_load_b128) so the DS scheduler can
// stagger s_wait_dscnt under the WMMA chain instead of waiting to 0 each time.
// ---------------------------------------------------------------------------

typedef __attribute__((ext_vector_type(16))) _Float16 v16h;
typedef __attribute__((ext_vector_type(8)))  _Float16 v8h;
typedef __attribute__((ext_vector_type(8)))  float    v8f;

#define HID    256
#define NPTS   4096      // P
#define STEPS  128
#define LDPAD  264       // padded LDS row stride in halves (528B/row -> bank shift of 4/row)
#define WVS    16        // waves per block (512 threads)

// workspace layout (bytes)
#define WS_T0     0
#define WS_SPAN   16384
#define WS_FBUF   49152                       // 524288 f32
#define WS_W2T    (49152 + 2097152)           // 65536 f16

// shared layout (bytes)
#define SM_W2T    0                           // 256*264 halves = 135168
#define SM_W1     (256*LDPAD*2)               // 768 f32
#define SM_B1     (SM_W1 + 768*4)             // 256 f32
#define SM_B2     (SM_B1 + 256*4)             // 256 f32
#define SM_W3     (SM_B2 + 256*4)             // 256 f32
#define SM_B3     (SM_W3 + 256*4)             // 4 f32 (pad)
#define SM_H1     (SM_B3 + 4*4)               // per-wave 16*LDPAD halves
#define SM_BYTES  (SM_H1 + WVS*16*LDPAD*2)    // = 276496 < 320KB/WGP

__device__ __forceinline__ float fast_tanh(float x) {
#if __has_builtin(__builtin_amdgcn_tanhf)
    return __builtin_amdgcn_tanhf(x);        // v_tanh_f32 (confirmed in disasm)
#else
    float cx = fminf(fmaxf(x, -15.0f), 15.0f);
    float e = __expf(2.0f * cx);
    return (e - 1.0f) / (e + 1.0f);
#endif
}

// Cooperative stage-in of weights: W2T (f16, padded rows), W1/b1/b2/W3/b3 (f32).
__device__ __forceinline__ void load_shared(char* smem,
                                            const float* W1, const float* b1,
                                            const _Float16* w2t,
                                            const float* b2, const float* W3,
                                            const float* b3) {
    _Float16* sW2T = (_Float16*)(smem + SM_W2T);
    float* sW1 = (float*)(smem + SM_W1);
    float* sB1 = (float*)(smem + SM_B1);
    float* sB2 = (float*)(smem + SM_B2);
    float* sW3 = (float*)(smem + SM_W3);
    float* sB3 = (float*)(smem + SM_B3);
    int tid = threadIdx.x;
    // 65536 halves as 8192 16-byte chunks
    for (int i = tid; i < 8192; i += blockDim.x) {
        int n  = i >> 5;           // 32 chunks per 256-half row
        int kc = (i & 31) << 3;
        *(v8h*)(sW2T + n * LDPAD + kc) = *(const v8h*)(w2t + n * HID + kc);
    }
    for (int i = tid; i < 768; i += blockDim.x) sW1[i] = W1[i];
    for (int i = tid; i < 256; i += blockDim.x) {
        sB1[i] = b1[i]; sB2[i] = b2[i]; sW3[i] = W3[i];
    }
    if (tid == 0) sB3[0] = b3[0];
    __syncthreads();
}

// Layer 1: this lane fills h1 row m = lane>>1, cols jbase..jbase+127,
// packed as v8h -> ds_store_b128 (16 stores instead of 128 b16 stores).
__device__ __forceinline__ void layer1(_Float16* sH1w, const float* sW1, const float* sB1,
                                       float px, float py, float pz, int lane) {
    int m = lane >> 1;
    int jbase = (lane & 1) * 128;
    _Float16* row = sH1w + m * LDPAD + jbase;
    for (int jc = 0; jc < 16; ++jc) {
        v8h pack;
#pragma unroll
        for (int u = 0; u < 8; ++u) {
            int jj = jbase + jc * 8 + u;
            float h = fmaf(px, sW1[jj],
                      fmaf(py, sW1[256 + jj],
                      fmaf(pz, sW1[512 + jj], sB1[jj])));
            pack[u] = (_Float16)fast_tanh(h);
        }
        *(v8h*)(row + jc * 8) = pack;
    }
}

// Layers 2+3 via WMMA. Assumes h1 tile (16 x 256, f16) already in sH1w and synced.
// Returns the field value for M = (lane&7) + (lane>=16 ? 8 : 0).
__device__ __forceinline__ float mlp_tail(const _Float16* sW2T, const _Float16* sH1w,
                                          const float* sB2, const float* sW3,
                                          float b3v, int lane) {
    const int  m16 = lane & 15;
    const bool hi  = (lane & 16) != 0;

    // A tiles (16x32 f16), ISA layout: lanes 0-15 hold K 0-7 & 16-23,
    // lanes 16-31 hold K 8-15 & 24-31. Cached for the whole tail (64 VGPRs).
    v16h a[8];
#pragma unroll
    for (int kt = 0; kt < 8; ++kt) {
        int kb = kt * 32 + (hi ? 8 : 0);
        v8h c0 = *(const v8h*)(sH1w + m16 * LDPAD + kb);
        v8h c1 = *(const v8h*)(sH1w + m16 * LDPAD + kb + 16);
        v16h av;
#pragma unroll
        for (int i = 0; i < 8; ++i) { av[i] = c0[i]; av[i + 8] = c1[i]; }
        a[kt] = av;
    }

    // B layout: lane = N column, same K split as A (lanes 0-15: K 0-7 & 16-23, 16-31: K 8-15 & 24-31).
    const _Float16* wrow = sW2T + m16 * LDPAD + (hi ? 8 : 0);

    v8f partial = {};
    for (int nt = 0; nt < 16; ++nt) {
        int n = nt * 16 + m16;
        // scalar epilogue operands issued first: their DS latency hides under WMMAs
        float bias = sB2[n], w3 = sW3[n];
        // batch the full N-tile of B (16x ds_load_b128) so waits can stagger
        v8h bb[16];
#pragma unroll
        for (int kt = 0; kt < 8; ++kt) {
            bb[2 * kt]     = *(const v8h*)(wrow + kt * 32);
            bb[2 * kt + 1] = *(const v8h*)(wrow + kt * 32 + 16);
        }
        v8f c = {};
#pragma unroll
        for (int kt = 0; kt < 8; ++kt) {
            v16h bv;
#pragma unroll
            for (int i = 0; i < 8; ++i) {
                bv[i]     = bb[2 * kt][i];
                bv[i + 8] = bb[2 * kt + 1][i];
            }
            c = __builtin_amdgcn_wmma_f32_16x16x32_f16(false, a[kt], false, bv,
                                                       (short)0, c, false, false);
        }
        wrow += 16 * LDPAD;
#pragma unroll
        for (int i = 0; i < 8; ++i)
            partial[i] += fast_tanh(c[i] + bias) * w3;   // layer 3 partial dot
    }

    // reduce over the 16 N-lanes of each half (C/D layout: lanes 0-15 => M 0-7, 16-31 => M 8-15)
#pragma unroll
    for (int mask = 1; mask < 16; mask <<= 1) {
#pragma unroll
        for (int i = 0; i < 8; ++i)
            partial[i] += __shfl_xor(partial[i], mask, 32);
    }
    int e = lane & 7;
    float val = partial[0];
#pragma unroll
    for (int i = 1; i < 8; ++i) val = (e == i) ? partial[i] : val;
    return val + b3v;
}

// --------------------------- kernel 0: W2 -> f16 transpose ------------------
__global__ void prep_w2_kernel(const float* __restrict__ W2, _Float16* __restrict__ w2t) {
    int n = blockIdx.x;      // output column-major row
    int k = threadIdx.x;
    w2t[n * HID + k] = (_Float16)W2[k * HID + n];
}

// --------------------------- kernel 1: cube slab setup ----------------------
__global__ void ray_setup_kernel(const float* __restrict__ cam_o, const float* __restrict__ dirs,
                                 float* __restrict__ rayT0, float* __restrict__ raySpan) {
    int r = blockIdx.x * blockDim.x + threadIdx.x;
    if (r >= NPTS) return;
    float tmin = -INFINITY, tmax = INFINITY;
#pragma unroll
    for (int c = 0; c < 3; ++c) {
        float oc = cam_o[r * 3 + c];
        float dc = dirs[r * 3 + c];
        dc = (dc == 0.0f) ? 1e-9f : dc;
        float inv = 1.0f / dc;
        float t1 = (-1.0f - oc) * inv;
        float t2 = ( 1.0f - oc) * inv;
        tmin = fmaxf(tmin, fminf(t1, t2));
        tmax = fminf(tmax, fmaxf(t1, t2));
    }
    bool valid = (tmax > tmin) && (tmax > 0.0f);
    float t0   = valid ? fmaxf(tmin, 0.0f) : 0.0f;   // fallback = linspace(0,4,128)
    float span = valid ? (tmax - t0)       : 4.0f;
    rayT0[r] = t0;
    raySpan[r] = span;
}

// --------------------------- kernel 2: field at all P*S points --------------
__global__ void __launch_bounds__(WVS * 32)
field_eval_kernel(const float* __restrict__ cam_o, const float* __restrict__ dirs,
                  const float* __restrict__ W1, const float* __restrict__ b1,
                  const _Float16* __restrict__ w2t,
                  const float* __restrict__ b2, const float* __restrict__ W3,
                  const float* __restrict__ b3,
                  const float* __restrict__ rayT0, const float* __restrict__ raySpan,
                  float* __restrict__ fbuf) {
    extern __shared__ char smem[];
    load_shared(smem, W1, b1, w2t, b2, W3, b3);
    const _Float16* sW2T = (const _Float16*)(smem + SM_W2T);
    const float* sW1 = (const float*)(smem + SM_W1);
    const float* sB1 = (const float*)(smem + SM_B1);
    const float* sB2 = (const float*)(smem + SM_B2);
    const float* sW3 = (const float*)(smem + SM_W3);
    const float  b3v = *(const float*)(smem + SM_B3);

    int lane = threadIdx.x & 31;
    int wave = threadIdx.x >> 5;
    _Float16* sH1w = (_Float16*)(smem + SM_H1) + wave * 16 * LDPAD;

    int waveGlobal = blockIdx.x * WVS + wave;        // 64 blocks * 16 waves = 1024
    // 32768 16-point groups, 32 per wave (uniform trip count -> __syncthreads legal)
    for (int i = 0; i < 32; ++i) {
        int g = waveGlobal * 32 + i;
        int r = g >> 3;                               // 8 groups per ray
        int sbase = (g & 7) * 16;
        float t0 = rayT0[r], span = raySpan[r];
        float ox = cam_o[r * 3], oy = cam_o[r * 3 + 1], oz = cam_o[r * 3 + 2];
        float dx = dirs[r * 3],  dy = dirs[r * 3 + 1], dz = dirs[r * 3 + 2];
        int m = lane >> 1;
        float t = t0 + span * (float)(sbase + m) * (1.0f / 127.0f);
        layer1(sH1w, sW1, sB1, ox + t * dx, oy + t * dy, oz + t * dz, lane);
        __syncthreads();
        float val = mlp_tail(sW2T, sH1w, sB2, sW3, b3v, lane);
        if ((lane & 8) == 0) {                        // one writer per M
            int M = (lane & 7) + ((lane & 16) ? 8 : 0);
            fbuf[r * STEPS + sbase + M] = val;
        }
        __syncthreads();
    }
}

// --------------------------- kernel 3: bracket scan + 8 secant steps --------
__global__ void __launch_bounds__(WVS * 32)
secant_kernel(const float* __restrict__ cam_o, const float* __restrict__ dirs,
              const float* __restrict__ W1, const float* __restrict__ b1,
              const _Float16* __restrict__ w2t,
              const float* __restrict__ b2, const float* __restrict__ W3,
              const float* __restrict__ b3,
              const float* __restrict__ rayT0, const float* __restrict__ raySpan,
              const float* __restrict__ fbuf, float* __restrict__ d_out) {
    extern __shared__ char smem[];
    load_shared(smem, W1, b1, w2t, b2, W3, b3);
    const _Float16* sW2T = (const _Float16*)(smem + SM_W2T);
    const float* sW1 = (const float*)(smem + SM_W1);
    const float* sB1 = (const float*)(smem + SM_B1);
    const float* sB2 = (const float*)(smem + SM_B2);
    const float* sW3 = (const float*)(smem + SM_W3);
    const float  b3v = *(const float*)(smem + SM_B3);

    int lane = threadIdx.x & 31;
    int wave = threadIdx.x >> 5;
    _Float16* sH1w = (_Float16*)(smem + SM_H1) + wave * 16 * LDPAD;

    int g  = blockIdx.x * WVS + wave;   // 16 blocks * 16 waves = 256 groups of 16 rays
    int r0 = g * 16;
    int myr = r0 + (lane & 15);

    // ----- bracket init on lanes 0..15 (one ray each) -----
    float x1 = 0.f, x2 = 0.f, f1 = 0.f, f2 = 0.f, xn = 0.f;
    int hit = 0;
    if (lane < 16) {
        const float* f = fbuf + myr * STEPS;
        float t0 = rayT0[myr], span = raySpan[myr];
        // argmax of sign-diff == first s with f[s]<0 && f[s+1]>0 (diff==2), else 127
        int index = STEPS - 1;
        float fprev = f[0];
        bool f0neg = fprev < 0.0f;
        for (int s = 0; s < STEPS - 1; ++s) {
            float fc = f[s + 1];
            if (index == STEPS - 1 && fprev < 0.0f && fc > 0.0f) index = s;
            fprev = fc;
        }
        hit = (index != STEPS - 1) && f0neg;
        int idx2 = min(index + 1, STEPS - 1);
        x1 = t0 + span * (float)index * (1.0f / 127.0f);
        x2 = t0 + span * (float)idx2  * (1.0f / 127.0f);
        f1 = f[index];
        f2 = f[idx2];
        float den = f2 - f1; if (den == 0.0f) den = 1e-8f;
        xn = (x1 * f2 - x2 * f1) / den;
    }

    // per-lane ray data for layer-1 row m = lane>>1
    int rm = r0 + (lane >> 1);
    float ox = cam_o[rm * 3], oy = cam_o[rm * 3 + 1], oz = cam_o[rm * 3 + 2];
    float dx = dirs[rm * 3],  dy = dirs[rm * 3 + 1], dz = dirs[rm * 3 + 2];

    for (int it = 0; it < 8; ++it) {
        float xm = __shfl(xn, lane >> 1, 32);         // broadcast x_n of row m
        layer1(sH1w, sW1, sB1, ox + xm * dx, oy + xm * dy, oz + xm * dz, lane);
        __syncthreads();
        float val = mlp_tail(sW2T, sH1w, sB2, sW3, b3v, lane);
        __syncthreads();
        // route f(M) back to owner lane: M=(lane&7)+((lane&16)?8:0)
        int src = (lane & 8) ? (lane + 8) : lane;
        float fn = __shfl(val, src, 32);
        if (lane < 16) {
            bool lower = fn < 0.0f;
            x1 = lower ? xn : x1;
            f1 = lower ? fn : f1;
            x2 = lower ? x2 : xn;
            f2 = lower ? f2 : fn;
            float den = f2 - f1; if (den == 0.0f) den = 1e-8f;
            xn = (x1 * f2 - x2 * f1) / den;
        }
    }
    if (lane < 16)
        d_out[myr] = hit ? xn : INFINITY;
}

// ---------------------------------------------------------------------------
extern "C" void kernel_launch(void* const* d_in, const int* in_sizes, int n_in,
                              void* d_out, int out_size, void* d_ws, size_t ws_size,
                              hipStream_t stream) {
    (void)in_sizes; (void)n_in; (void)out_size; (void)ws_size;
    const float* cam_o = (const float*)d_in[0];
    const float* dirs  = (const float*)d_in[1];
    const float* W1    = (const float*)d_in[2];
    const float* b1    = (const float*)d_in[3];
    const float* W2    = (const float*)d_in[4];
    const float* b2    = (const float*)d_in[5];
    const float* W3    = (const float*)d_in[6];
    const float* b3    = (const float*)d_in[7];
    float* out = (float*)d_out;

    char* ws = (char*)d_ws;
    float*    rayT0   = (float*)(ws + WS_T0);
    float*    raySpan = (float*)(ws + WS_SPAN);
    float*    fbuf    = (float*)(ws + WS_FBUF);
    _Float16* w2t     = (_Float16*)(ws + WS_W2T);

    prep_w2_kernel<<<HID, HID, 0, stream>>>(W2, w2t);
    ray_setup_kernel<<<NPTS / 256, 256, 0, stream>>>(cam_o, dirs, rayT0, raySpan);
    field_eval_kernel<<<64, WVS * 32, SM_BYTES, stream>>>(cam_o, dirs, W1, b1, w2t, b2, W3, b3,
                                                          rayT0, raySpan, fbuf);
    secant_kernel<<<16, WVS * 32, SM_BYTES, stream>>>(cam_o, dirs, W1, b1, w2t, b2, W3, b3,
                                                      rayT0, raySpan, fbuf, out);
}
